// FlexMoERouter_3435973837301
// MI455X (gfx1250) — matrix operand order repaired and checked by reading, not compiled, not run on hardware
//
#include <hip/hip_runtime.h>
#include <hip/hip_bf16.h>

// ---------------------------------------------------------------------------
// FlexMoE router for MI455X (gfx1250, wave32, WMMA).
// - bf16 WMMA 16x16x32 GEMMs, f32 accumulation.
// - Weights transpose-converted to [N,K]: all LDS staging is b128 copies,
//   all fragment reads are contiguous-K (ds_load_b128 pairs).
// - Double-buffered LDS: 1 barrier per K-step, next tile loaded to registers
//   under the 4 WMMAs of the current step.
// - dispatch/combine outputs zero-filled then sparsely scattered
//   (reference only ever writes capacity slot 0).
// ---------------------------------------------------------------------------

typedef __attribute__((ext_vector_type(16))) __bf16 bf16x16;
typedef __attribute__((ext_vector_type(8)))  float  v8f;

#define NTOK  2048          // B*S
#define HDIM  2048
#define H2DIM 1024
#define EDIM  16
#define TOPK  4
#define CAP   768
#define NMOD  4

static __device__ __forceinline__ unsigned short f32_to_bf16(float f) {
  unsigned int u = __float_as_uint(f);
  u += 0x7fffu + ((u >> 16) & 1u);          // round-to-nearest-even
  return (unsigned short)(u >> 16);
}

// ---------------------------------------------------------------- zero fill
__global__ __launch_bounds__(256) void zero_f32(float* __restrict__ p, long n) {
  long i = (long)blockIdx.x * blockDim.x + threadIdx.x;
  long stride = (long)gridDim.x * blockDim.x;
  long n4 = n >> 2;
  float4* p4 = (float4*)p;
  float4 z = make_float4(0.f, 0.f, 0.f, 0.f);
  for (long j = i; j < n4; j += stride) p4[j] = z;
  for (long j = (n4 << 2) + i; j < n; j += stride) p[j] = 0.f;
}

// ------------------------------------------------- f32 -> bf16 (row-major)
__global__ __launch_bounds__(256) void cvt_bf16(const float* __restrict__ s,
                                                unsigned short* __restrict__ d,
                                                long n) {
  long i = (long)blockIdx.x * blockDim.x + threadIdx.x;
  long stride = (long)gridDim.x * blockDim.x;
  for (long j = i; j < n; j += stride) d[j] = f32_to_bf16(s[j]);
}

// --------------------------------- f32 [R,C] -> bf16 [C,R] (LDS tiled 32x32)
__global__ __launch_bounds__(256) void cvt_bf16_T(const float* __restrict__ s,
                                                  unsigned short* __restrict__ d,
                                                  int R, int C) {
  __shared__ float tile[32][33];
  const int r0 = blockIdx.y * 32, c0 = blockIdx.x * 32;
  const int tx = threadIdx.x & 31, ty = threadIdx.x >> 5;   // 32 x 8
#pragma unroll
  for (int i = 0; i < 32; i += 8) {
    int r = r0 + ty + i, c = c0 + tx;
    tile[ty + i][tx] = (r < R && c < C) ? s[(long)r * C + c] : 0.f;
  }
  __syncthreads();
#pragma unroll
  for (int i = 0; i < 32; i += 8) {
    int c = c0 + ty + i, r = r0 + tx;
    if (c < C && r < R) d[(long)c * R + r] = f32_to_bf16(tile[tx][ty + i]);
  }
}

// ------------------------------------------------------ tiled WMMA bf16 GEMM
// C[M,N] = act(A[M,K] * Bt[N,K]^T + bias[N]).
// Block tile 128x64 (BK=32), 8 waves in 4x2; each wave owns a 32x32 tile
// (2x2 WMMA 16x16x32). Double-buffered LDS, one barrier per K-step.
#define BM 128
#define BN 64
#define BK 32

__global__ __launch_bounds__(256) void gemm_bf16_wmma(
    const unsigned short* __restrict__ A,     // [M,K]  bf16 row-major
    const unsigned short* __restrict__ Bt,    // [N,K]  bf16 row-major (B^T)
    const float* __restrict__ bias,           // [N]
    unsigned short* __restrict__ Cbf,         // optional bf16 out [M,N]
    float* __restrict__ Cf,                   // optional f32 out  [M,N]
    int M, int N, int Kd, int relu) {
  __shared__ __align__(16) unsigned short As[2][BM * BK];   // 2 x 8 KiB
  __shared__ __align__(16) unsigned short Bs[2][BN * BK];   // 2 x 4 KiB

  const int tid  = threadIdx.x;
  const int lane = tid & 31;
  const int wave = tid >> 5;      // 0..7
  const int wm   = wave >> 1;     // 0..3 -> 32-row band
  const int wn   = wave & 1;      // 0..1 -> 32-col band
  const int m0   = blockIdx.y * BM;
  const int n0   = blockIdx.x * BN;

  v8f acc[2][2] = {};

  // staging map: A 128x32 -> 2 consecutive b128/thread (same row),
  //              Bt 64x32 -> 1 b128/thread
  const int arow = tid >> 1;                    // 0..127
  const int aq   = ((tid * 2) & 3) * 8;         // 0 or 16 (ushorts)
  const int brow = tid >> 2;                    // 0..63
  const int bq   = (tid & 3) * 8;               // 0,8,16,24

  const long aoff = (long)(m0 + arow) * Kd + aq;
  const long boff = (long)(n0 + brow) * Kd + bq;

  const int kb8  = (lane < 16) ? 0 : 8;
  const int kb16 = (lane < 16) ? 0 : 16;

  // ---- prologue: stage tile 0 ----
  {
    uint4 ra0 = *(const uint4*)(A + aoff);
    uint4 ra1 = *(const uint4*)(A + aoff + 8);
    uint4 rb  = *(const uint4*)(Bt + boff);
    *(uint4*)(As[0] + arow * BK + aq)     = ra0;
    *(uint4*)(As[0] + arow * BK + aq + 8) = ra1;
    *(uint4*)(Bs[0] + brow * BK + bq)     = rb;
  }
  __syncthreads();

  const int nIter = Kd / BK;
  for (int it = 0; it < nIter; ++it) {
    const int cur = it & 1;
    const bool hasNext = (it + 1) < nIter;

    // issue next tile's global loads early; they overlap the WMMAs below
    uint4 na0 = {}, na1 = {}, nb = {};
    if (hasNext) {
      const long ka = aoff + (long)(it + 1) * BK;
      const long kb = boff + (long)(it + 1) * BK;
      na0 = *(const uint4*)(A + ka);
      na1 = *(const uint4*)(A + ka + 8);
      nb  = *(const uint4*)(Bt + kb);
      if (it + 2 < nIter) {
        __builtin_prefetch(A + aoff + (long)(it + 2) * BK, 0, 1);
        __builtin_prefetch(Bt + boff + (long)(it + 2) * BK, 0, 1);
      }
    }

    // ---- fragments (contiguous-K: lane half h holds k = 16h..16h+15) ----
    union U { unsigned int u[8]; bf16x16 v; } af[2], bfr[2];
#pragma unroll
    for (int mi = 0; mi < 2; ++mi) {
      int rl = wm * 32 + mi * 16 + (lane & 15);
#pragma unroll
      for (int i = 0; i < 8; ++i) {
        int kk = kb16 + 2 * i;
        af[mi].u[i] = *(const unsigned int*)(As[cur] + rl * BK + kk);
      }
    }
#pragma unroll
    for (int ni = 0; ni < 2; ++ni) {
      int cl = wn * 32 + ni * 16 + (lane & 15);
#pragma unroll
      for (int i = 0; i < 8; ++i)
        bfr[ni].u[i] = *(const unsigned int*)(Bs[cur] + cl * BK + kb16 + 2 * i);
    }
#pragma unroll
    for (int mi = 0; mi < 2; ++mi)
#pragma unroll
      for (int ni = 0; ni < 2; ++ni)
        acc[mi][ni] = __builtin_amdgcn_wmma_f32_16x16x32_bf16(
            false, af[mi].v, false, bfr[ni].v, (short)0, acc[mi][ni],
            false, false);

    if (hasNext) {
      const int nxt = cur ^ 1;
      *(uint4*)(As[nxt] + arow * BK + aq)     = na0;
      *(uint4*)(As[nxt] + arow * BK + aq + 8) = na1;
      *(uint4*)(Bs[nxt] + brow * BK + bq)     = nb;
      __syncthreads();        // single barrier per K-step
    }
  }

  // ---- epilogue: C VGPR r holds row r (lanes 0-15) / r+8 (lanes 16-31) ----
#pragma unroll
  for (int mi = 0; mi < 2; ++mi) {
    const int rbase = m0 + wm * 32 + mi * 16 + ((lane < 16) ? 0 : 8);
#pragma unroll
    for (int ni = 0; ni < 2; ++ni) {
      const int colg  = n0 + wn * 32 + ni * 16 + (lane & 15);
      const float bcl = bias ? bias[colg] : 0.f;
#pragma unroll
      for (int r = 0; r < 8; ++r) {
        float v = acc[mi][ni][r] + bcl;
        if (relu) v = v > 0.f ? v : 0.f;
        long idx = (long)(rbase + r) * N + colg;
        if (Cf)  Cf[idx]  = v;
        if (Cbf) Cbf[idx] = f32_to_bf16(v);
      }
    }
  }
}

// ------------------------------------------- WMMA GEMM for N=16 logit heads
// Bt is [16,K] (transposed) -> fully contiguous-K fragment loads from global.
__global__ __launch_bounds__(256) void gemm_bf16_wmma_n16(
    const unsigned short* __restrict__ A,     // [M,K] bf16
    const unsigned short* __restrict__ Bt,    // [16,K] bf16
    const float* __restrict__ bias,           // [16]
    float* __restrict__ Cf,                   // [M,16] f32
    int M, int Kd) {
  const int tid  = threadIdx.x;
  const int lane = tid & 31;
  const int wave = tid >> 5;
  const int row0 = (blockIdx.x * 8 + wave) * 16;
  if (row0 >= M) return;

  const int rl   = row0 + (lane & 15);
  const int kb8  = (lane < 16) ? 0 : 8;
  const int col  = lane & 15;
  const int kb16 = (lane < 16) ? 0 : 16;

  v8f acc = {};
  for (int k0 = 0; k0 < Kd; k0 += 32) {
    union U { unsigned int u[8]; bf16x16 v; } af, bf;
#pragma unroll
    for (int i = 0; i < 8; ++i) {
      int kk = kb16 + 2 * i;
      af.u[i] = *(const unsigned int*)(A + (long)rl * Kd + k0 + kk);
    }
#pragma unroll
    for (int i = 0; i < 8; ++i)
      bf.u[i] = *(const unsigned int*)(Bt + (long)col * Kd + k0 + kb16 + 2 * i);
    acc = __builtin_amdgcn_wmma_f32_16x16x32_bf16(
        false, af.v, false, bf.v, (short)0, acc, false, false);
  }
  const float bcl = bias[col];
  const int rbase = row0 + ((lane < 16) ? 0 : 8);
#pragma unroll
  for (int r = 0; r < 8; ++r)
    Cf[(long)(rbase + r) * EDIM + col] = acc[r] + bcl;
}

// -------------------------------------------------- per-token router finalize
__global__ __launch_bounds__(256) void router_finalize(
    const float* __restrict__ g_logits,       // [NTOK,16]
    const float* __restrict__ s_logits,       // [4,NTOK,16]
    float* __restrict__ out,                  // full d_out
    float* __restrict__ expert_sum) {         // [16]
  const long COMB_OFF = (long)NTOK * EDIM * CAP;
  const long RP_OFF   = 2 * COMB_OFF;
  int t = blockIdx.x * blockDim.x + threadIdx.x;
  if (t >= NTOK) return;

  float rp[EDIM];
  {  // generalized router softmax, weight ALPHA = 0.7
    float l[EDIM], mx = -1e30f, sum = 0.f;
#pragma unroll
    for (int e = 0; e < EDIM; ++e) { l[e] = g_logits[(long)t * EDIM + e]; mx = fmaxf(mx, l[e]); }
#pragma unroll
    for (int e = 0; e < EDIM; ++e) { l[e] = expf(l[e] - mx); sum += l[e]; }
    float inv = 0.7f / sum;
#pragma unroll
    for (int e = 0; e < EDIM; ++e) rp[e] = l[e] * inv;
  }
  // 4 specialized softmaxes; each contributes (1-ALPHA)/NUM_AVAILABLE = 0.15
  for (int m = 0; m < NMOD; ++m) {
    const float* base = s_logits + ((long)m * NTOK + t) * EDIM;
    float l[EDIM], mx = -1e30f, sum = 0.f;
#pragma unroll
    for (int e = 0; e < EDIM; ++e) { l[e] = base[e]; mx = fmaxf(mx, l[e]); }
#pragma unroll
    for (int e = 0; e < EDIM; ++e) { l[e] = expf(l[e] - mx); sum += l[e]; }
    float inv = 0.15f / sum;
#pragma unroll
    for (int e = 0; e < EDIM; ++e) rp[e] += l[e] * inv;
  }

#pragma unroll
  for (int e = 0; e < EDIM; ++e) {
    out[RP_OFF + (long)t * EDIM + e] = rp[e];
    atomicAdd(&expert_sum[e], rp[e]);
  }

  // top-4 of 16, renormalize, scatter into capacity slot 0
  float tmp[EDIM];
#pragma unroll
  for (int e = 0; e < EDIM; ++e) tmp[e] = rp[e];
  int   idxs[TOPK];
  float vals[TOPK];
  float tsum = 0.f;
  for (int j = 0; j < TOPK; ++j) {
    int bi = 0; float bv = tmp[0];
    for (int e = 1; e < EDIM; ++e) if (tmp[e] > bv) { bv = tmp[e]; bi = e; }
    idxs[j] = bi; vals[j] = bv; tsum += bv; tmp[bi] = -1e30f;
  }
  float invt = 1.f / tsum;
  for (int j = 0; j < TOPK; ++j) {
    long base = ((long)t * EDIM + idxs[j]) * CAP;   // capacity slot 0
    out[base]            = 1.0f;                    // dispatch
    out[COMB_OFF + base] = vals[j] * invt;          // combine
  }
}

// ---------------------------------------------------------------- aux loss
__global__ void aux_loss_k(const float* __restrict__ expert_sum,
                           float* __restrict__ out) {
  if (blockIdx.x == 0 && threadIdx.x == 0) {
    const long AUX_OFF = 2L * NTOK * EDIM * CAP + (long)NTOK * EDIM;
    float a = 0.f;
    for (int e = 0; e < EDIM; ++e) {
      float m = expert_sum[e] / (float)NTOK;
      a += m * logf(m * (float)EDIM + 1e-9f);
    }
    out[AUX_OFF] = a;
  }
}

// ---------------------------------------------------------------------------
extern "C" void kernel_launch(void* const* d_in, const int* in_sizes, int n_in,
                              void* d_out, int out_size, void* d_ws, size_t ws_size,
                              hipStream_t stream) {
  (void)in_sizes; (void)n_in; (void)out_size; (void)ws_size;
  const float* hidden = (const float*)d_in[0];
  const float* img    = (const float*)d_in[1];
  const float* gen    = (const float*)d_in[2];
  const float* Wg1    = (const float*)d_in[3];
  const float* bg1    = (const float*)d_in[4];
  const float* Wg2    = (const float*)d_in[5];
  const float* bg2    = (const float*)d_in[6];
  const float* Wm1    = (const float*)d_in[7];
  const float* bm1    = (const float*)d_in[8];
  const float* Wm2    = (const float*)d_in[9];
  const float* bm2    = (const float*)d_in[10];
  const float* Ws1    = (const float*)d_in[11];
  const float* bs1    = (const float*)d_in[12];
  const float* Ws2    = (const float*)d_in[13];
  const float* bs2    = (const float*)d_in[14];
  float* out = (float*)d_out;

  // ---- workspace layout (bytes); *_t weight buffers hold B^T [N,K] ----
  unsigned char* w = (unsigned char*)d_ws;
  const size_t SZ_HH = (size_t)NTOK * HDIM * 2;          // 8 MiB
  unsigned short* hid_bf  = (unsigned short*)(w);
  unsigned short* img_bf  = (unsigned short*)(w + 1 * SZ_HH);
  unsigned short* gen_bf  = (unsigned short*)(w + 2 * SZ_HH);
  unsigned short* wm1_t   = (unsigned short*)(w + 3 * SZ_HH);
  unsigned short* wm2_t   = (unsigned short*)(w + 4 * SZ_HH);
  unsigned short* wg1_t   = (unsigned short*)(w + 5 * SZ_HH);
  unsigned short* wg2_t   = (unsigned short*)(w + 6 * SZ_HH);                       // 64 KiB
  unsigned short* ws1_t   = (unsigned short*)(w + 6 * SZ_HH + (64u << 10));         // 16 MiB
  unsigned short* ws2_t   = (unsigned short*)(w + 8 * SZ_HH + (64u << 10));         // 128 KiB
  unsigned short* t1_bf   = (unsigned short*)(w + 8 * SZ_HH + (192u << 10));        // relu(hid@Wm1)
  unsigned short* miss_bf = (unsigned short*)(w + 9 * SZ_HH + (192u << 10));
  unsigned short* gh_bf   = (unsigned short*)(w + 10 * SZ_HH + (192u << 10));
  unsigned short* sh_bf   = (unsigned short*)(w + 11 * SZ_HH + (192u << 10));       // [4][2048,1024]
  float* g_log  = (float*)(w + 13 * SZ_HH + (192u << 10));                          // [2048,16]
  float* s_log  = (float*)(w + 13 * SZ_HH + (320u << 10));                          // [4,2048,16]
  float* e_sum  = (float*)(w + 13 * SZ_HH + (832u << 10));                          // [16]

  // 1) zero dispatch+combine region of d_out and expert accumulator
  zero_f32<<<1024, 256, 0, stream>>>(out, 2L * NTOK * EDIM * CAP);
  zero_f32<<<1, 64, 0, stream>>>(e_sum, 16);

  // 2) activations: plain convert; weights: transpose-convert to [N,K]
  cvt_bf16<<<1024, 256, 0, stream>>>(hidden, hid_bf, (long)NTOK * HDIM);
  cvt_bf16<<<1024, 256, 0, stream>>>(img,    img_bf, (long)NTOK * HDIM);
  cvt_bf16<<<1024, 256, 0, stream>>>(gen,    gen_bf, (long)NTOK * HDIM);
  {
    dim3 gT(HDIM / 32, HDIM / 32);
    cvt_bf16_T<<<gT, 256, 0, stream>>>(Wm1, wm1_t, HDIM, HDIM);
    cvt_bf16_T<<<gT, 256, 0, stream>>>(Wm2, wm2_t, HDIM, HDIM);
    cvt_bf16_T<<<gT, 256, 0, stream>>>(Wg1, wg1_t, HDIM, HDIM);
  }
  cvt_bf16_T<<<dim3(1, HDIM / 32), 256, 0, stream>>>(Wg2, wg2_t, HDIM, EDIM);
  for (int m = 0; m < NMOD; ++m) {
    cvt_bf16_T<<<dim3(H2DIM / 32, HDIM / 32), 256, 0, stream>>>(
        Ws1 + (size_t)m * HDIM * H2DIM, ws1_t + (size_t)m * HDIM * H2DIM,
        HDIM, H2DIM);
    cvt_bf16_T<<<dim3(1, H2DIM / 32), 256, 0, stream>>>(
        Ws2 + (size_t)m * H2DIM * EDIM, ws2_t + (size_t)m * H2DIM * EDIM,
        H2DIM, EDIM);
  }

  // 3) big WMMA GEMMs (block tile 128x64, double-buffered LDS)
  dim3 blk(256);
  dim3 gHH(HDIM / BN, NTOK / BM);     // (32, 16)
  dim3 gH2(H2DIM / BN, NTOK / BM);    // (16, 16)
  gemm_bf16_wmma<<<gHH, blk, 0, stream>>>(hid_bf, wm1_t, bm1, t1_bf, nullptr,
                                          NTOK, HDIM, HDIM, 1);
  gemm_bf16_wmma<<<gHH, blk, 0, stream>>>(t1_bf, wm2_t, bm2, miss_bf, nullptr,
                                          NTOK, HDIM, HDIM, 0);
  gemm_bf16_wmma<<<gHH, blk, 0, stream>>>(hid_bf, wg1_t, bg1, gh_bf, nullptr,
                                          NTOK, HDIM, HDIM, 1);
  const unsigned short* mod_in[NMOD] = { img_bf, gen_bf, miss_bf, miss_bf };
  for (int m = 0; m < NMOD; ++m) {
    gemm_bf16_wmma<<<gH2, blk, 0, stream>>>(
        mod_in[m], ws1_t + (size_t)m * HDIM * H2DIM, bs1 + (size_t)m * H2DIM,
        sh_bf + (size_t)m * NTOK * H2DIM, nullptr, NTOK, H2DIM, HDIM, 1);
  }

  // 4) logit heads (N = 16, Bt = [16,K])
  gemm_bf16_wmma_n16<<<NTOK / 128, blk, 0, stream>>>(gh_bf, wg2_t, bg2, g_log,
                                                     NTOK, HDIM);
  for (int m = 0; m < NMOD; ++m) {
    gemm_bf16_wmma_n16<<<NTOK / 128, blk, 0, stream>>>(
        sh_bf + (size_t)m * NTOK * H2DIM, ws2_t + (size_t)m * H2DIM * EDIM,
        bs2 + (size_t)m * EDIM, s_log + (size_t)m * NTOK * EDIM, NTOK, H2DIM);
  }

  // 5) softmax blend + top-k + scatter; 6) aux loss
  router_finalize<<<NTOK / 256, blk, 0, stream>>>(g_log, s_log, out, e_sum);
  aux_loss_k<<<1, 64, 0, stream>>>(e_sum, out);
}
